// Model_8546984919511
// MI455X (gfx1250) — matrix operand order repaired and checked by reading, not compiled, not run on hardware
//
#include <hip/hip_runtime.h>
#include <hip/hip_bf16.h>

// ---------------------------------------------------------------------------
// R-GCN on MI455X (gfx1250): edge-grouped bf16 WMMA GEMMs + f32 atomic scatter.
// h_bf (25MB) + agg (51MB) are L2-resident (192MB L2); edges sorted by etype
// on-device so each 16-edge WMMA tile shares one relation matrix B.
// Register-blocked: one wave computes the full 16 x dout output row-block,
// reusing each gathered A fragment across all column tiles (8 WMMAs).
// Pad edges scatter into a dummy agg row (row N) -> unconditional atomics;
// all epilogue addressing is 32-bit so atomics/loads use SADDR+voffset form.
// ---------------------------------------------------------------------------

typedef __bf16 bf16;
typedef __attribute__((ext_vector_type(16))) __bf16 v16bf;
typedef __attribute__((ext_vector_type(8)))  float  v8f;

static constexpr int WPB = 8;   // waves (tiles) per block
static constexpr int RMAX = 12;

// meta layout (ints): [0..15] hist, [16..31] cursor, [32..48] offs, [49] numTiles
#define M_HIST 0
#define M_CUR  16
#define M_OFF  32
#define M_NT   49

__global__ void k_zero_meta(int* meta) {
    if (threadIdx.x < 64) meta[threadIdx.x] = 0;
}

__global__ void k_hist(const int* __restrict__ et, int* __restrict__ meta, int E) {
    int e = blockIdx.x * 256 + threadIdx.x;
    if (e < E) atomicAdd(&meta[M_HIST + et[e]], 1);
}

__global__ void k_scan(int* meta, int R) {
    if (threadIdx.x == 0 && blockIdx.x == 0) {
        int off = 0;
        for (int r = 0; r < R; ++r) {
            meta[M_OFF + r] = off;
            meta[M_CUR + r] = off;
            int c = meta[M_HIST + r];
            off += (c + 15) & ~15;          // pad each relation segment to 16
        }
        meta[M_OFF + R] = off;
        meta[M_NT] = off >> 4;
    }
}

__global__ void k_scatter(const int* __restrict__ src, const int* __restrict__ dst,
                          const int* __restrict__ et, int* __restrict__ meta,
                          int* __restrict__ srcS, int* __restrict__ dstS, int E) {
    int e = blockIdx.x * 256 + threadIdx.x;
    if (e >= E) return;
    int r = et[e];
    int pos = atomicAdd(&meta[M_CUR + r], 1);
    srcS[pos] = src[e];
    dstS[pos] = dst[e];
}

// pads: src = N (zero sentinel feature row), dst = N (dummy agg row)
__global__ void k_pad(const int* __restrict__ meta, int* __restrict__ srcS,
                      int* __restrict__ dstS, int N) {
    int r = blockIdx.x;                    // R blocks x 16 threads
    int start = meta[M_OFF + r] + meta[M_HIST + r];
    int end   = meta[M_OFF + r + 1];
    int p = start + (int)threadIdx.x;
    if (p < end) { srcS[p] = N; dstS[p] = N; }
}

__global__ void k_tilerel(const int* __restrict__ meta, int* __restrict__ tileRel,
                          int R, int Tmax) {
    int t = blockIdx.x * 256 + threadIdx.x;
    if (t >= Tmax || t >= meta[M_NT]) return;
    int e = t << 4, r = 0;
    for (int q = 1; q < R; ++q)
        if (e >= meta[M_OFF + q]) r = q;
    tileRel[t] = r;
}

// fp32 [N,16] -> bf16 [N+1,32] (cols 16..31 zero, row N zero sentinel)
__global__ void k_convert(const float* __restrict__ h, bf16* __restrict__ out, int N) {
    int id = blockIdx.x * 256 + threadIdx.x;
    int total = (N + 1) * 32;
    if (id >= total) return;
    int n = id >> 5, c = id & 31;
    float v = (n < N && c < 16) ? h[n * 16 + c] : 0.f;
    out[id] = (bf16)v;
}

// Pack W [R,din,dout] fp32 into per-lane WMMA B-fragment layout (bf16):
// idx = (((r*nkc + kc)*nnt + nt)*32 + lane)*16 + j ; k = kc*32 + (lane>=16?8:0) + (j<8?j:8+j)
__global__ void k_packW(const float* __restrict__ W, bf16* __restrict__ out,
                        int R, int din, int dout, int dinP) {
    int nkc = dinP >> 5, nnt = dout >> 4;
    int total = R * nkc * nnt * 512;
    int id = blockIdx.x * 256 + threadIdx.x;
    if (id >= total) return;
    int j    = id & 15;
    int lane = (id >> 4) & 31;
    int rest = id >> 9;
    int nt = rest % nnt; rest /= nnt;
    int kc = rest % nkc;
    int r  = rest / nkc;
    int k = kc * 32 + ((lane >= 16) ? 8 : 0) + ((j < 8) ? j : 8 + j);
    int n = nt * 16 + (lane & 15);
    float v = (k < din) ? W[((size_t)r * din + k) * dout + n] : 0.f;
    out[id] = (bf16)v;
}

// ---------------------------------------------------------------------------
// Edge message GEMM: one wave = one 16-edge tile x full dout (NNT col tiles).
// A (16x32 bf16) gathered rows of h_bf (loaded ONCE per k-step, reused by NNT
// WMMAs), B from packed W[rel], D scattered into agg via f32 atomics
// (unconditional; pads target dummy row N with exactly-zero messages).
// ---------------------------------------------------------------------------
template <int NKC, int NNT>
__global__ void __launch_bounds__(256)
k_edge(const bf16* __restrict__ hin, int strideIn,
       const bf16* __restrict__ Wpk,
       const int* __restrict__ srcS, const int* __restrict__ dstS,
       const int* __restrict__ tileRel, const int* __restrict__ meta,
       float* __restrict__ agg) {
    constexpr int DOUT = NNT * 16;
    int lane = threadIdx.x & 31, wave = threadIdx.x >> 5;
    int tile = blockIdx.x * WPB + wave;
    if (tile >= meta[M_NT]) return;
    int r  = tileRel[tile];
    int e0 = tile << 4;

    int srcRow = srcS[e0 + (lane & 15)];
    const bf16* aRow  = hin + (size_t)srcRow * strideIn + ((lane >= 16) ? 8 : 0);
    const bf16* bBase = Wpk + (size_t)r * (NKC * NNT * 512) + lane * 16;

    v8f zero = {};
    v8f acc[NNT];
#pragma unroll
    for (int nt = 0; nt < NNT; ++nt) acc[nt] = zero;

#pragma unroll
    for (int kc = 0; kc < NKC; ++kc) {
        union { v16bf v; uint4 q[2]; } A;
        const bf16* ap = aRow + kc * 32;
        A.q[0] = *(const uint4*)(ap);          // k = base+0..7
        A.q[1] = *(const uint4*)(ap + 16);     // k = base+16..23
#pragma unroll
        for (int nt = 0; nt < NNT; ++nt) {
            union { v16bf v; uint4 q[2]; } Bf;
            const bf16* bp = bBase + ((size_t)(kc * NNT + nt)) * 512;
            Bf.q[0] = *(const uint4*)(bp);
            Bf.q[1] = *(const uint4*)(bp + 8);
            acc[nt] = __builtin_amdgcn_wmma_f32_16x16x32_bf16(
                false, A.v, false, Bf.v, (short)0, acc[nt], false, false);
        }
    }

    unsigned ro   = (lane >= 16) ? 8u : 0u;
    unsigned col0 = (unsigned)(lane & 15);
    // 8 consecutive dst indices -> two b128 loads
    int4 d0 = *(const int4*)(dstS + e0 + ro);
    int4 d1 = *(const int4*)(dstS + e0 + ro + 4);
    int dests[8] = {d0.x, d0.y, d0.z, d0.w, d1.x, d1.y, d1.z, d1.w};

    // 32-bit row offsets (agg < 4GB) -> SADDR + voffset atomics
    unsigned rowOff[8];
#pragma unroll
    for (int i = 0; i < 8; ++i)
        rowOff[i] = (unsigned)dests[i] * (unsigned)DOUT + col0;

#pragma unroll
    for (int i = 0; i < 8; ++i) {
#pragma unroll
        for (int nt = 0; nt < NNT; ++nt) {
            atomicAdd(agg + rowOff[i] + (unsigned)(nt * 16), acc[nt][i]);
        }
    }
}

// ---------------------------------------------------------------------------
// Self-loop GEMM fused with epilogue: h_next = relu(agg + h@L + b) -> bf16.
// One wave = 16 nodes x full dout.
// ---------------------------------------------------------------------------
template <int NKC, int NNT>
__global__ void __launch_bounds__(256)
k_node(const bf16* __restrict__ hin, int strideIn,
       const bf16* __restrict__ Lpk,
       const float* __restrict__ agg, const float* __restrict__ bias,
       bf16* __restrict__ hout, int N) {
    constexpr int DOUT = NNT * 16;
    int lane = threadIdx.x & 31, wave = threadIdx.x >> 5;
    int tile = blockIdx.x * WPB + wave;
    int n0 = tile << 4;
    if (n0 >= N) return;

    int ar = n0 + (lane & 15); if (ar > N) ar = N;
    const bf16* aRow  = hin + (size_t)ar * strideIn + ((lane >= 16) ? 8 : 0);
    const bf16* bBase = Lpk + lane * 16;

    v8f zero = {};
    v8f acc[NNT];
#pragma unroll
    for (int nt = 0; nt < NNT; ++nt) acc[nt] = zero;

#pragma unroll
    for (int kc = 0; kc < NKC; ++kc) {
        union { v16bf v; uint4 q[2]; } A;
        const bf16* ap = aRow + kc * 32;
        A.q[0] = *(const uint4*)(ap);
        A.q[1] = *(const uint4*)(ap + 16);
#pragma unroll
        for (int nt = 0; nt < NNT; ++nt) {
            union { v16bf v; uint4 q[2]; } Bf;
            const bf16* bp = bBase + ((size_t)(kc * NNT + nt)) * 512;
            Bf.q[0] = *(const uint4*)(bp);
            Bf.q[1] = *(const uint4*)(bp + 8);
            acc[nt] = __builtin_amdgcn_wmma_f32_16x16x32_bf16(
                false, A.v, false, Bf.v, (short)0, acc[nt], false, false);
        }
    }

    unsigned ro   = (lane >= 16) ? 8u : 0u;
    unsigned col0 = (unsigned)(lane & 15);
#pragma unroll
    for (int i = 0; i < 8; ++i) {
        int row = n0 + (int)ro + i;
        if (row < N) {
            unsigned base = (unsigned)row * (unsigned)DOUT + col0;
#pragma unroll
            for (int nt = 0; nt < NNT; ++nt) {
                unsigned idx = base + (unsigned)(nt * 16);
                float v = acc[nt][i] + agg[idx] + bias[(unsigned)(nt * 16) + col0];
                v = v > 0.f ? v : 0.f;
                hout[idx] = (bf16)v;
            }
        }
    }
}

__global__ void k_zero_row(bf16* p, int n) {
    int i = blockIdx.x * 256 + threadIdx.x;
    if (i < n) p[i] = (bf16)0.f;
}

// z = h4 @ Wd + bd  (64 -> 2)
__global__ void k_dense(const bf16* __restrict__ h4, const float* __restrict__ Wd,
                        const float* __restrict__ bd, float* __restrict__ z, int N) {
    int n = blockIdx.x * 256 + threadIdx.x;
    if (n >= N) return;
    float a0 = bd[0], a1 = bd[1];
    const bf16* row = h4 + (size_t)n * 64;
#pragma unroll 8
    for (int k = 0; k < 64; ++k) {
        float v = (float)row[k];
        a0 += v * Wd[2 * k];
        a1 += v * Wd[2 * k + 1];
    }
    z[2 * n] = a0;
    z[2 * n + 1] = a1;
}

__global__ void k_pairs(const int* __restrict__ ei, const float* __restrict__ z,
                        float* __restrict__ out, int P) {
    int p = blockIdx.x * 256 + threadIdx.x;
    if (p >= P) return;
    int u1 = ei[4 * p], v1 = ei[4 * p + 1], u2 = ei[4 * p + 2], v2 = ei[4 * p + 3];
    out[2 * p]             = 0.5f * (z[2 * u1]     + z[2 * v1]);
    out[2 * p + 1]         = 0.5f * (z[2 * u1 + 1] + z[2 * v1 + 1]);
    out[2 * P + 2 * p]     = 0.5f * (z[2 * u2]     + z[2 * v2]);
    out[2 * P + 2 * p + 1] = 0.5f * (z[2 * u2 + 1] + z[2 * v2 + 1]);
}

extern "C" void kernel_launch(void* const* d_in, const int* in_sizes, int n_in,
                              void* d_out, int out_size, void* d_ws, size_t ws_size,
                              hipStream_t stream) {
    const float* h   = (const float*)d_in[0];
    const int* src   = (const int*)d_in[1];
    const int* dst   = (const int*)d_in[2];
    const int* ety   = (const int*)d_in[3];
    const int* ei    = (const int*)d_in[4];
    const float* W[4] = {(const float*)d_in[5],  (const float*)d_in[8],
                         (const float*)d_in[11], (const float*)d_in[14]};
    const float* L[4] = {(const float*)d_in[6],  (const float*)d_in[9],
                         (const float*)d_in[12], (const float*)d_in[15]};
    const float* B[4] = {(const float*)d_in[7],  (const float*)d_in[10],
                         (const float*)d_in[13], (const float*)d_in[16]};
    const float* Wd = (const float*)d_in[17];
    const float* bd = (const float*)d_in[18];

    const int N = in_sizes[0] / 16;
    const int E = in_sizes[1];
    const int P = in_sizes[4] / 4;
    const int R = RMAX;
    const int Tmax = (E + R * 16) / 16;   // worst-case padded tile count

    const int din_[4]  = {16, 128, 128, 128};
    const int dinP[4]  = {32, 128, 128, 128};
    const int dout_[4] = {128, 128, 128, 64};

    // ---- carve workspace -------------------------------------------------
    char* wp = (char*)d_ws;
    auto carve = [&](size_t bytes) -> void* {
        void* p = (void*)wp;
        wp += (bytes + 255) & ~(size_t)255;
        return p;
    };
    int*  meta    = (int*)carve(64 * sizeof(int));
    bf16* hA      = (bf16*)carve((size_t)(N + 1) * 128 * sizeof(bf16));
    bf16* hB      = (bf16*)carve((size_t)(N + 1) * 128 * sizeof(bf16));
    float* agg    = (float*)carve((size_t)(N + 1) * 128 * sizeof(float)); // +dummy row N
    int*  srcS    = (int*)carve((size_t)(E + R * 16) * sizeof(int));
    int*  dstS    = (int*)carve((size_t)(E + R * 16) * sizeof(int));
    int*  tileRel = (int*)carve((size_t)Tmax * sizeof(int));
    bf16* Wpk[4]; bf16* Lpk[4];
    for (int li = 0; li < 4; ++li) {
        size_t we = (size_t)R * (dinP[li] >> 5) * (dout_[li] >> 4) * 512;
        size_t le = (size_t)(dinP[li] >> 5) * (dout_[li] >> 4) * 512;
        Wpk[li] = (bf16*)carve(we * sizeof(bf16));
        Lpk[li] = (bf16*)carve(le * sizeof(bf16));
    }
    float* z = (float*)carve((size_t)N * 2 * sizeof(float));

    // ---- sort edges by relation (counting sort, padded to 16) ------------
    k_zero_meta<<<1, 64, 0, stream>>>(meta);
    k_hist<<<(E + 255) / 256, 256, 0, stream>>>(ety, meta, E);
    k_scan<<<1, 32, 0, stream>>>(meta, R);
    k_scatter<<<(E + 255) / 256, 256, 0, stream>>>(src, dst, ety, meta, srcS, dstS, E);
    k_pad<<<R, 16, 0, stream>>>(meta, srcS, dstS, N);
    k_tilerel<<<(Tmax + 255) / 256, 256, 0, stream>>>(meta, tileRel, R, Tmax);

    // ---- pack weights to WMMA fragment layout ----------------------------
    for (int li = 0; li < 4; ++li) {
        int we = R * (dinP[li] >> 5) * (dout_[li] >> 4) * 512;
        int le = (dinP[li] >> 5) * (dout_[li] >> 4) * 512;
        k_packW<<<(we + 255) / 256, 256, 0, stream>>>(W[li], Wpk[li], R, din_[li], dout_[li], dinP[li]);
        k_packW<<<(le + 255) / 256, 256, 0, stream>>>(L[li], Lpk[li], 1, din_[li], dout_[li], dinP[li]);
    }

    // ---- input features -> bf16, padded to K=32, with zero sentinel row --
    k_convert<<<((N + 1) * 32 + 255) / 256, 256, 0, stream>>>(h, hA, N);

    // ---- 4 R-GCN layers --------------------------------------------------
    bf16* hin = hA; bf16* hout = hB;
    int nodeTiles = (N + 15) / 16;
    dim3 gE((Tmax + WPB - 1) / WPB);
    dim3 gN((nodeTiles + WPB - 1) / WPB);
    for (int li = 0; li < 4; ++li) {
        int dout = dout_[li];
        hipMemsetAsync(agg, 0, (size_t)(N + 1) * dout * sizeof(float), stream);

        if (li == 0) {
            k_edge<1, 8><<<gE, 32 * WPB, 0, stream>>>(hin, dinP[li], Wpk[li], srcS, dstS, tileRel, meta, agg);
            k_node<1, 8><<<gN, 32 * WPB, 0, stream>>>(hin, dinP[li], Lpk[li], agg, B[li], hout, N);
        } else if (li == 3) {
            k_edge<4, 4><<<gE, 32 * WPB, 0, stream>>>(hin, dinP[li], Wpk[li], srcS, dstS, tileRel, meta, agg);
            k_node<4, 4><<<gN, 32 * WPB, 0, stream>>>(hin, dinP[li], Lpk[li], agg, B[li], hout, N);
        } else {
            k_edge<4, 8><<<gE, 32 * WPB, 0, stream>>>(hin, dinP[li], Wpk[li], srcS, dstS, tileRel, meta, agg);
            k_node<4, 8><<<gN, 32 * WPB, 0, stream>>>(hin, dinP[li], Lpk[li], agg, B[li], hout, N);
        }
        k_zero_row<<<1, 128, 0, stream>>>(hout + (size_t)N * dout, dout);

        bf16* t = hin; hin = hout; hout = t;
    }

    // ---- dense head + edge-pair scoring ----------------------------------
    k_dense<<<(N + 255) / 256, 256, 0, stream>>>(hin, Wd, bd, z, N);
    k_pairs<<<(P + 255) / 256, 256, 0, stream>>>(ei, z, (float*)d_out, P);
    (void)n_in; (void)out_size; (void)ws_size;
}
